// FractalAttention3D_4784593568517
// MI455X (gfx1250) — compile-verified, matrix operand order checked by reading, not run on hardware
//
#include <hip/hip_runtime.h>
#include <hip/hip_bf16.h>

typedef __attribute__((ext_vector_type(16))) __bf16 v16bf;
typedef __attribute__((ext_vector_type(8)))  float  v8f;
typedef __attribute__((ext_vector_type(4))) unsigned int u32x4;
typedef __attribute__((ext_vector_type(8))) int         i32x8;
typedef __attribute__((ext_vector_type(4))) int         i32x4;
typedef unsigned short u16;
typedef unsigned int   u32;

#define R_BLOCKS 13824
#define D_BLOCKS 1728
#define EDIM 512

__device__ __forceinline__ u16 f32_to_bf16(float f) {
    u32 u = __float_as_uint(f);
    u32 r = u + 0x7FFFu + ((u >> 16) & 1u);   // round-to-nearest-even
    return (u16)(r >> 16);
}

// Load a 16x32 bf16 A-style fragment from a row-major matrix (leading dim in
// elements).  ISA layout (CDNA5 ISA, 16-bit A 16x32):
//   lanes 0-15 : row M=lane,    VGPR0-3 = K 0..7,  VGPR4-7 = K 16..23
//   lanes 16-31: row M=lane-16, VGPR0-3 = K 8..15, VGPR4-7 = K 24..31
// The B operand of D=A*B uses the same per-lane layout applied to B^T, so
// this loader also produces B fragments from row-major [N x K] matrices.
__device__ __forceinline__ v16bf load_frag_rm(const u16* base, int ldim) {
    int lane = threadIdx.x & 31;
    int row  = lane & 15;
    int koff = (lane >> 4) << 3;          // 0 or 8
    const u16* p = base + (size_t)row * ldim + koff;
    union { v16bf v; u32 w[8]; } u;
    const u32* p0 = (const u32*)p;
    const u32* p1 = (const u32*)(p + 16);
    u.w[0] = p0[0]; u.w[1] = p0[1]; u.w[2] = p0[2]; u.w[3] = p0[3];
    u.w[4] = p1[0]; u.w[5] = p1[1]; u.w[6] = p1[2]; u.w[7] = p1[3];
    return u.v;
}

__device__ __forceinline__ v8f bwmma(v16bf a, v16bf b, v8f c) {
    return __builtin_amdgcn_wmma_f32_16x16x32_bf16(
        false, a, false, b, (short)0, c, false, false);
}

// ---------------- prep kernels ----------------

__global__ void add_convert_kernel(const float* __restrict__ a,
                                   const float* __restrict__ b,
                                   u16* __restrict__ outp, int n) {
    int i = blockIdx.x * blockDim.x + threadIdx.x;
    if (i < n) outp[i] = f32_to_bf16(a[i] + b[i]);
}

// Pack [W_p(256); W_c(128); W_o(128)] (each [*,512] row-major) into one
// 512x512 bf16 matrix + one 512 fp32 bias vector.
__global__ void pack_w_kernel(const float* __restrict__ Wp, const float* __restrict__ bp,
                              const float* __restrict__ Wc, const float* __restrict__ bc,
                              const float* __restrict__ Wo, const float* __restrict__ bo,
                              u16* __restrict__ W, float* __restrict__ bias) {
    int i = blockIdx.x * blockDim.x + threadIdx.x;
    if (i >= 512 * 512) return;
    int row = i >> 9, col = i & 511;
    float w;
    if (row < 256)      w = Wp[row * 512 + col];
    else if (row < 384) w = Wc[(row - 256) * 512 + col];
    else                w = Wo[(row - 384) * 512 + col];
    W[i] = f32_to_bf16(w);
    if (col == 0)
        bias[row] = (row < 256) ? bp[row] : (row < 384 ? bc[row - 256] : bo[row - 384]);
}

// pooled_domains [D,64] f32 -> dfT [64,D] bf16 (so PV B-fragments are contiguous)
__global__ void transpose_df_kernel(const float* __restrict__ pooled,
                                    u16* __restrict__ dfT) {
    int i = blockIdx.x * blockDim.x + threadIdx.x;
    if (i >= D_BLOCKS * 64) return;
    int d = i >> 6, j = i & 63;
    dfT[(size_t)j * D_BLOCKS + d] = f32_to_bf16(pooled[i]);
}

// ---------------- projection GEMM: C[M,512] = A[M,512] @ W^T + bias ----------------
// One wave per 16x16 output tile.  grid = (M/16, 32), block = 32.
__global__ void gemm_proj_kernel(const u16* __restrict__ A,
                                 const u16* __restrict__ W,
                                 const float* __restrict__ bias,
                                 u16* __restrict__ C) {
    int tm = blockIdx.x * 16;
    int tn = blockIdx.y * 16;
    v8f acc = {};
    #pragma unroll
    for (int k = 0; k < 512; k += 32) {
        v16bf a = load_frag_rm(A + (size_t)tm * 512 + k, 512);
        v16bf b = load_frag_rm(W + (size_t)tn * 512 + k, 512);
        acc = bwmma(a, b, acc);
    }
    int lane  = threadIdx.x & 31;
    int col   = lane & 15;
    int rbase = (lane >> 4) << 3;
    float bcol = bias[tn + col];
    #pragma unroll
    for (int v = 0; v < 8; ++v) {
        float x = acc[v] + bcol;
        C[(size_t)(tm + rbase + v) * 512 + tn + col] = f32_to_bf16(x);
    }
}

// ---------------- fused flash-attention-style kernel ----------------
// One wave handles 16 R-rows; streams D in tiles of 32 with online softmax.
// rproj/dproj column layout: [0,256)=repr, [256,384)=contrast, [384,512)=offset.
__global__ void fused_attn_kernel(const u16* __restrict__ rproj,
                                  const u16* __restrict__ dproj,
                                  const u16* __restrict__ dfT,
                                  float* __restrict__ out) {
    __shared__ u16 ldsQ[16 * 520];   // 16 rows x 512, +16B/row pad (bank conflicts)
    __shared__ u16 ldsP[16 * 40];    // 16 x 32 P*contrast tile, padded

    const int lane = threadIdx.x;
    const int r0   = blockIdx.x * 16;

    // --- Stage this wave's 16 query rows (16 x 1024 B tile) into LDS via the
    // Tensor Data Mover.  The TDM's LDS-padding feature inserts 4 DWORDs
    // (16 B) after every 256 DWORDs (1024 B = one row), reproducing the
    // 520-element padded row stride in hardware.  (D# layout per
    // CDNA5 ISA; 6-arg clang builtin form.)
    {
        unsigned long long gaddr =
            (unsigned long long)(uintptr_t)(rproj + (size_t)r0 * 512);
        u32 lds_base = (u32)(uintptr_t)ldsQ;   // flat->LDS: low 32 bits = offset
        u32x4 g0 = {
            1u,                                             // count=1, user mode
            lds_base,                                       // lds_addr
            (u32)gaddr,                                     // global_addr[31:0]
            ((u32)(gaddr >> 32) & 0x01FFFFFFu) | (2u << 30) // addr[56:32] | type=2
        };
        i32x8 g1 = {
            (int)((1u << 16) | (1u << 20) | (7u << 22) | (3u << 25)),
            // ^ data_size=2B, pad_enable, pad_interval=256 DW, pad_amount=4 DW
            (int)(512u << 16),   // tensor_dim0 = 512 (bits 63:48)
            (int)(16u << 16),    // tensor_dim0 hi=0 | tensor_dim1 = 16
            (int)(512u << 16),   // tensor_dim1 hi=0 | tile_dim0 = 512
            16,                  // tile_dim1 = 16 | tile_dim2 = 0
            512,                 // tensor_dim0_stride = 512
            0, 0                 // dim0_stride hi | tensor_dim1_stride = 0
        };
        i32x4 gz4 = { 0, 0, 0, 0 };
        i32x8 gz8 = { 0, 0, 0, 0, 0, 0, 0, 0 };
        __builtin_amdgcn_tensor_load_to_lds(g0, g1, gz4, gz4, gz8, 0);
        __builtin_amdgcn_s_wait_tensorcnt((short)0);
    }
    __syncthreads();

    v8f acc0 = {}, acc1 = {}, acc2 = {}, acc3 = {};
    float m_run[8], l_run[8], off_run[8];
    #pragma unroll
    for (int v = 0; v < 8; ++v) { m_run[v] = -1e30f; l_run[v] = 0.0f; off_run[v] = 0.0f; }

    const float scale_s    = 0.25f;                // 1/(sqrt(256)*0.25)
    const float inv_sqrt_h = 0.08838834764831845f; // 1/sqrt(128)

    for (int d0 = 0; d0 < D_BLOCKS; d0 += 32) {
        v8f s0 = {}, s1 = {}, c0 = {}, c1 = {}, o0 = {}, o1 = {};
        const u16* dp0 = dproj + (size_t)d0 * 512;
        const u16* dp1 = dproj + (size_t)(d0 + 16) * 512;

        #pragma unroll
        for (int k = 0; k < 256; k += 32) {        // logits, K=256
            v16bf a  = load_frag_rm(ldsQ + k, 520);
            v16bf b0 = load_frag_rm(dp0 + k, 512);
            v16bf b1 = load_frag_rm(dp1 + k, 512);
            s0 = bwmma(a, b0, s0);
            s1 = bwmma(a, b1, s1);
        }
        #pragma unroll
        for (int k = 0; k < 128; k += 32) {        // contrast & offset, K=128 each
            v16bf a  = load_frag_rm(ldsQ + 256 + k, 520);
            v16bf b0 = load_frag_rm(dp0 + 256 + k, 512);
            v16bf b1 = load_frag_rm(dp1 + 256 + k, 512);
            c0 = bwmma(a, b0, c0);
            c1 = bwmma(a, b1, c1);
            a  = load_frag_rm(ldsQ + 384 + k, 520);
            b0 = load_frag_rm(dp0 + 384 + k, 512);
            b1 = load_frag_rm(dp1 + 384 + k, 512);
            o0 = bwmma(a, b0, o0);
            o1 = bwmma(a, b1, o1);
        }

        // Online softmax + gating.  C-fragment layout: lane holds rows
        // M = v + 8*(lane>=16) for column N = lane&15; row reductions run
        // across the 16 lanes of each half-wave.
        int col  = lane & 15;
        int half = lane >> 4;
        #pragma unroll
        for (int v = 0; v < 8; ++v) {
            float sv0 = s0[v] * scale_s;
            float sv1 = s1[v] * scale_s;
            float mx  = fmaxf(sv0, sv1);
            #pragma unroll
            for (int off = 1; off < 16; off <<= 1)
                mx = fmaxf(mx, __shfl_xor(mx, off, 16));
            float mnew  = fmaxf(m_run[v], mx);
            float alpha = __expf(m_run[v] - mnew);
            m_run[v] = mnew;

            float e0 = __expf(sv0 - mnew);
            float e1 = __expf(sv1 - mnew);
            float cv0 = tanhf(c0[v] * inv_sqrt_h) * 1.8f;
            float cv1 = tanhf(c1[v] * inv_sqrt_h) * 1.8f;
            float ov0 = tanhf(o0[v] * inv_sqrt_h);
            float ov1 = tanhf(o1[v] * inv_sqrt_h);

            float ls = e0 + e1;
            float os = e0 * ov0 + e1 * ov1;
            #pragma unroll
            for (int off = 1; off < 16; off <<= 1) {
                ls += __shfl_xor(ls, off, 16);
                os += __shfl_xor(os, off, 16);
            }
            l_run[v]   = l_run[v]   * alpha + ls;
            off_run[v] = off_run[v] * alpha + os;

            int mrow = v + (half << 3);
            ldsP[mrow * 40 + col]      = f32_to_bf16(e0 * cv0);
            ldsP[mrow * 40 + 16 + col] = f32_to_bf16(e1 * cv1);

            acc0[v] *= alpha; acc1[v] *= alpha; acc2[v] *= alpha; acc3[v] *= alpha;
        }
        __syncthreads();

        // PV: (16x32 P*contrast) @ (32 x 64 domain_flat tile), via dfT rows.
        v16bf pa = load_frag_rm(ldsP, 40);
        v16bf b;
        b = load_frag_rm(dfT + (size_t)(0 * 16) * D_BLOCKS + d0, D_BLOCKS); acc0 = bwmma(pa, b, acc0);
        b = load_frag_rm(dfT + (size_t)(1 * 16) * D_BLOCKS + d0, D_BLOCKS); acc1 = bwmma(pa, b, acc1);
        b = load_frag_rm(dfT + (size_t)(2 * 16) * D_BLOCKS + d0, D_BLOCKS); acc2 = bwmma(pa, b, acc2);
        b = load_frag_rm(dfT + (size_t)(3 * 16) * D_BLOCKS + d0, D_BLOCKS); acc3 = bwmma(pa, b, acc3);
        __syncthreads();
    }

    // Epilogue: out = acc/l + off/l, 64 cols per row.
    int col   = lane & 15;
    int rbase = (lane >> 4) << 3;
    #pragma unroll
    for (int v = 0; v < 8; ++v) {
        float invl = 1.0f / l_run[v];
        float offv = off_run[v] * invl;
        size_t base = (size_t)(r0 + rbase + v) * 64;
        out[base +  0 + col] = acc0[v] * invl + offv;
        out[base + 16 + col] = acc1[v] * invl + offv;
        out[base + 32 + col] = acc2[v] * invl + offv;
        out[base + 48 + col] = acc3[v] * invl + offv;
    }
}

// ---------------- launch ----------------

extern "C" void kernel_launch(void* const* d_in, const int* in_sizes, int n_in,
                              void* d_out, int out_size, void* d_ws, size_t ws_size,
                              hipStream_t stream) {
    const float* pooled = (const float*)d_in[0];
    const float* rpos   = (const float*)d_in[1];
    const float* dpos   = (const float*)d_in[2];
    const float* rlat   = (const float*)d_in[3];
    const float* dlat   = (const float*)d_in[4];
    const float* W_rp = (const float*)d_in[5];  const float* b_rp = (const float*)d_in[6];
    const float* W_dp = (const float*)d_in[7];  const float* b_dp = (const float*)d_in[8];
    const float* W_rc = (const float*)d_in[9];  const float* b_rc = (const float*)d_in[10];
    const float* W_dc = (const float*)d_in[11]; const float* b_dc = (const float*)d_in[12];
    const float* W_ro = (const float*)d_in[13]; const float* b_ro = (const float*)d_in[14];
    const float* W_do = (const float*)d_in[15]; const float* b_do = (const float*)d_in[16];
    float* out = (float*)d_out;

    // Workspace carve-up (bf16 buffers stored as u16). ~33 MB total.
    u16* rb    = (u16*)d_ws;                       // R*512  range base bf16
    u16* db    = rb    + (size_t)R_BLOCKS * 512;   // D*512  domain base bf16
    u16* wr    = db    + (size_t)D_BLOCKS * 512;   // 512*512 packed range weights
    u16* wd    = wr    + (size_t)512 * 512;        // 512*512 packed domain weights
    u16* rproj = wd    + (size_t)512 * 512;        // R*512  range projections bf16
    u16* dproj = rproj + (size_t)R_BLOCKS * 512;   // D*512  domain projections bf16
    u16* dft   = dproj + (size_t)D_BLOCKS * 512;   // 64*D   domain_flat^T bf16
    float* br  = (float*)(dft + (size_t)64 * D_BLOCKS); // 512 range bias
    float* bd  = br + 512;                              // 512 domain bias

    {
        int n = R_BLOCKS * 512;
        add_convert_kernel<<<(n + 255) / 256, 256, 0, stream>>>(rlat, rpos, rb, n);
    }
    {
        int n = D_BLOCKS * 512;
        add_convert_kernel<<<(n + 255) / 256, 256, 0, stream>>>(dlat, dpos, db, n);
    }
    pack_w_kernel<<<(512 * 512 + 255) / 256, 256, 0, stream>>>(
        W_rp, b_rp, W_rc, b_rc, W_ro, b_ro, wr, br);
    pack_w_kernel<<<(512 * 512 + 255) / 256, 256, 0, stream>>>(
        W_dp, b_dp, W_dc, b_dc, W_do, b_do, wd, bd);
    transpose_df_kernel<<<(D_BLOCKS * 64 + 255) / 256, 256, 0, stream>>>(pooled, dft);

    gemm_proj_kernel<<<dim3(R_BLOCKS / 16, 32), 32, 0, stream>>>(rb, wr, br, rproj);
    gemm_proj_kernel<<<dim3(D_BLOCKS / 16, 32), 32, 0, stream>>>(db, wd, bd, dproj);

    fused_attn_kernel<<<R_BLOCKS / 16, 32, 0, stream>>>(rproj, dproj, dft, out);
}
